// MultiHeadAttention_59871844106290
// MI455X (gfx1250) — matrix operand order repaired and checked by reading, not compiled, not run on hardware
//
#include <hip/hip_runtime.h>

// ---- CDNA5 / gfx1250 WMMA types -------------------------------------------
typedef __attribute__((ext_vector_type(16))) __bf16 v16bf;
typedef __attribute__((ext_vector_type(8)))  __bf16 v8bf;
typedef __attribute__((ext_vector_type(8)))  float  v8f;

#define N_EMBD 1024
#define N_HEAD 16
#define HEAD   64
#define SEQ    2048
#define BATCH  2
#define MROWS  (BATCH * SEQ)   // 4096

__device__ __forceinline__ __bf16 f2bf(float f) {
    unsigned u = __builtin_bit_cast(unsigned, f);
    u += 0x7FFFu + ((u >> 16) & 1u);           // round-to-nearest-even
    unsigned short h = (unsigned short)(u >> 16);
    return __builtin_bit_cast(__bf16, h);
}

__device__ __forceinline__ v16bf cat8(v8bf lo, v8bf hi) {
    return __builtin_shufflevector(lo, hi, 0,1,2,3,4,5,6,7,8,9,10,11,12,13,14,15);
}

__device__ __forceinline__ v8f wmma_bf16(v16bf a, v16bf b, v8f c) {
    // D = A(16x32 bf16) * B(32x16 bf16) + C(16x16 f32)
    return __builtin_amdgcn_wmma_f32_16x16x32_bf16(false, a, false, b, (short)0, c,
                                                   false, false);
}

// ---- fp32 -> bf16 conversion ----------------------------------------------
__global__ __launch_bounds__(256) void cvt_bf16(const float* __restrict__ src,
                                                __bf16* __restrict__ dst, int n) {
    int base = (blockIdx.x * 256 + threadIdx.x) * 8;
    #pragma unroll
    for (int i = 0; i < 8; ++i) {
        int idx = base + i;
        if (idx < n) dst[idx] = f2bf(src[idx]);
    }
}

// ---- bf16 WMMA GEMM: out = A[M,1024] @ W[N=1024,K=1024]^T ------------------
// Wave tile: 32 rows x 64 cols (2 A-fragments, 8 f32 accumulators).
// mode 0: bf16 out in [B,H,T,d]      (Q, K)
// mode 1: bf16 out in [B,H,d,T]      (V transposed for the PV matmul)
// mode 2: f32  out row-major + bias  (final projection)
__global__ __launch_bounds__(256) void gemm1024(const __bf16* __restrict__ A,
                                                const __bf16* __restrict__ W,
                                                void* __restrict__ outp,
                                                const float* __restrict__ bias,
                                                int mode) {
    const int lane = threadIdx.x & 31;
    const int w    = threadIdx.x >> 5;      // 8 waves / block
    const int ln   = lane & 15;
    const int half = lane >> 4;

    const int m0 = blockIdx.x * 256 + w * 32;   // 32 rows per wave
    const int n0 = blockIdx.y * 64;             // 64 cols per wave

    v8f acc[2][4];
    #pragma unroll
    for (int rh = 0; rh < 2; ++rh)
        #pragma unroll
        for (int nt = 0; nt < 4; ++nt)
            #pragma unroll
            for (int i = 0; i < 8; ++i) acc[rh][nt][i] = 0.0f;

    const __bf16* arow0 = A + (size_t)(m0 + ln) * N_EMBD;        // rows m0..m0+15
    const __bf16* arow1 = A + (size_t)(m0 + 16 + ln) * N_EMBD;   // rows m0+16..m0+31
    for (int k0 = 0; k0 < N_EMBD; k0 += 32) {
        v16bf aL = cat8(*(const v8bf*)(arow0 + k0 + half * 8),
                        *(const v8bf*)(arow0 + k0 + 16 + half * 8));
        v16bf aH = cat8(*(const v8bf*)(arow1 + k0 + half * 8),
                        *(const v8bf*)(arow1 + k0 + 16 + half * 8));
        __builtin_prefetch(arow0 + k0 + 64, 0, 1);   // global_prefetch_b8 (speculative)
        __builtin_prefetch(arow1 + k0 + 64, 0, 1);
        #pragma unroll
        for (int nt = 0; nt < 4; ++nt) {
            const __bf16* wrow = W + (size_t)(n0 + nt * 16 + ln) * N_EMBD
                                   + k0 + half * 8;
            v16bf b = cat8(*(const v8bf*)wrow, *(const v8bf*)(wrow + 16));
            __builtin_prefetch(wrow + 64, 0, 1);
            acc[0][nt] = wmma_bf16(aL, b, acc[0][nt]);
            acc[1][nt] = wmma_bf16(aH, b, acc[1][nt]);
        }
    }

    if (mode == 2) {
        float* out = (float*)outp;
        #pragma unroll
        for (int rh = 0; rh < 2; ++rh)
            #pragma unroll
            for (int nt = 0; nt < 4; ++nt)
                #pragma unroll
                for (int v = 0; v < 8; ++v) {
                    int m = m0 + rh * 16 + v + 8 * half;
                    int n = n0 + nt * 16 + ln;
                    out[(size_t)m * N_EMBD + n] = acc[rh][nt][v] + bias[n];
                }
    } else {
        __bf16* out = (__bf16*)outp;
        #pragma unroll
        for (int rh = 0; rh < 2; ++rh)
            #pragma unroll
            for (int nt = 0; nt < 4; ++nt)
                #pragma unroll
                for (int v = 0; v < 8; ++v) {
                    int m = m0 + rh * 16 + v + 8 * half;  // global row -> (b, t)
                    int n = n0 + nt * 16 + ln;            // global col -> (h, dd)
                    int b  = m >> 11, t = m & 2047;
                    int h  = n >> 6,  dd = n & 63;
                    size_t idx = (mode == 0)
                        ? (((size_t)(b * N_HEAD + h) * SEQ + t) * HEAD + dd)
                        : (((size_t)(b * N_HEAD + h) * HEAD + dd) * SEQ + t);
                    out[idx] = f2bf(acc[rh][nt][v]);
                }
    }
}

// ---- causal flash attention: one wave per 16-query tile --------------------
// Q,K: [B,H,T,64] bf16; Vt: [B,H,64,T] bf16; O: [B,T,1024] bf16
__global__ __launch_bounds__(256) void attn_flash(const __bf16* __restrict__ Q,
                                                  const __bf16* __restrict__ K,
                                                  const __bf16* __restrict__ Vt,
                                                  __bf16* __restrict__ O) {
    __shared__ __bf16 plds[8 * 16 * 32];        // per-wave 16x32 P transpose patch

    const int lane = threadIdx.x & 31;
    const int w    = threadIdx.x >> 5;
    const int ln   = lane & 15;
    const int half = lane >> 4;

    const int bh = blockIdx.y;                  // b*16 + h
    const int b  = bh >> 4;
    const int h  = bh & 15;
    const int q0 = (blockIdx.x * 8 + w) * 16;

    const __bf16* Qh = Q  + (size_t)bh * SEQ * HEAD;
    const __bf16* Kh = K  + (size_t)bh * SEQ * HEAD;
    const __bf16* Vh = Vt + (size_t)bh * HEAD * SEQ;

    // Q A-fragments (16x32 each, covering d=0..31 and d=32..63)
    const __bf16* qrow = Qh + (size_t)(q0 + ln) * HEAD;
    v16bf qa0 = cat8(*(const v8bf*)(qrow +      half * 16),
                     *(const v8bf*)(qrow + 8 +  half * 16));
    v16bf qa1 = cat8(*(const v8bf*)(qrow + 32 + half * 16),
                     *(const v8bf*)(qrow + 40 + half * 16));

    float m_r[8], l_r[8];
    v8f acc[4];
    #pragma unroll
    for (int v = 0; v < 8; ++v) { m_r[v] = -1.0e30f; l_r[v] = 0.0f; }
    #pragma unroll
    for (int nt = 0; nt < 4; ++nt)
        #pragma unroll
        for (int i = 0; i < 8; ++i) acc[nt][i] = 0.0f;

    const float scale = 0.125f;                 // 1/sqrt(64)
    const int kend = ((q0 + 47) >> 5) << 5;     // key blocks covering keys <= q0+15

    for (int kb = 0; kb < kend; kb += 32) {
        // ---- S = Q @ K^T for 32 keys (two 16-key N-subtiles) ----
        v8f s0, s1;
        #pragma unroll
        for (int i = 0; i < 8; ++i) { s0[i] = 0.0f; s1[i] = 0.0f; }

        const __bf16* kr0 = Kh + (size_t)(kb + ln)      * HEAD + half * 16;
        const __bf16* kr1 = Kh + (size_t)(kb + 16 + ln) * HEAD + half * 16;
        __builtin_prefetch(kr0 + 32 * HEAD, 0, 1);   // next key block (speculative)
        s0 = wmma_bf16(qa0, *(const v16bf*)(kr0),      s0);
        s0 = wmma_bf16(qa1, *(const v16bf*)(kr0 + 32), s0);
        s1 = wmma_bf16(qa0, *(const v16bf*)(kr1),      s1);
        s1 = wmma_bf16(qa1, *(const v16bf*)(kr1 + 32), s1);

        #pragma unroll
        for (int v = 0; v < 8; ++v) { s0[v] *= scale; s1[v] *= scale; }

        if (kb + 31 > q0) {                     // partial (diagonal) block: mask
            #pragma unroll
            for (int v = 0; v < 8; ++v) {
                int mg = q0 + v + 8 * half;
                if (kb + ln      > mg) s0[v] = -1.0e30f;
                if (kb + 16 + ln > mg) s1[v] = -1.0e30f;
            }
        }

        // ---- online softmax (row reductions within 16-lane halves) ----
        float fac[8];
        #pragma unroll
        for (int v = 0; v < 8; ++v) {
            float t = fmaxf(s0[v], s1[v]);
            t = fmaxf(t, __shfl_xor(t, 1));
            t = fmaxf(t, __shfl_xor(t, 2));
            t = fmaxf(t, __shfl_xor(t, 4));
            t = fmaxf(t, __shfl_xor(t, 8));
            float mn = fmaxf(m_r[v], t);
            fac[v] = __expf(m_r[v] - mn);
            m_r[v] = mn;
            s0[v] = __expf(s0[v] - mn);
            s1[v] = __expf(s1[v] - mn);
            float rs = s0[v] + s1[v];
            rs += __shfl_xor(rs, 1);
            rs += __shfl_xor(rs, 2);
            rs += __shfl_xor(rs, 4);
            rs += __shfl_xor(rs, 8);
            l_r[v] = l_r[v] * fac[v] + rs;
        }
        #pragma unroll
        for (int nt = 0; nt < 4; ++nt)
            #pragma unroll
            for (int v = 0; v < 8; ++v) acc[nt][v] *= fac[v];

        // ---- P: C-layout -> A-layout via per-wave LDS patch ----
        __bf16* pl = plds + w * 512;            // 16 rows x 32 cols
        #pragma unroll
        for (int v = 0; v < 8; ++v) {
            int row = v + 8 * half;
            pl[row * 32 + ln]      = f2bf(s0[v]);
            pl[row * 32 + 16 + ln] = f2bf(s1[v]);
        }
        asm volatile("s_wait_dscnt 0" ::: "memory");
        const __bf16* prow = pl + ln * 32;
        v16bf pa = cat8(*(const v8bf*)(prow +      half * 16),
                        *(const v8bf*)(prow + 8 +  half * 16));

        // ---- O += P @ V  (Vt rows give contiguous key-dim per lane) ----
        #pragma unroll
        for (int nt = 0; nt < 4; ++nt) {
            const __bf16* vrow = Vh + (size_t)(nt * 16 + ln) * SEQ + kb + half * 16;
            __builtin_prefetch(vrow + 32, 0, 1);
            acc[nt] = wmma_bf16(pa, *(const v16bf*)vrow, acc[nt]);
        }
    }

    // ---- epilogue: normalize, write O in [B,T,C] for the projection GEMM ----
    #pragma unroll
    for (int nt = 0; nt < 4; ++nt)
        #pragma unroll
        for (int v = 0; v < 8; ++v) {
            int trow = q0 + v + 8 * half;
            int col  = h * HEAD + nt * 16 + ln;
            O[((size_t)b * SEQ + trow) * N_EMBD + col] = f2bf(acc[nt][v] / l_r[v]);
        }
}

// ---- launcher --------------------------------------------------------------
extern "C" void kernel_launch(void* const* d_in, const int* in_sizes, int n_in,
                              void* d_out, int out_size, void* d_ws, size_t ws_size,
                              hipStream_t stream) {
    const float* x  = (const float*)d_in[0];
    const float* Wq = (const float*)d_in[1];
    const float* Wk = (const float*)d_in[2];
    const float* Wv = (const float*)d_in[3];
    const float* Wp = (const float*)d_in[4];
    const float* bp = (const float*)d_in[5];

    const size_t eX = (size_t)MROWS * N_EMBD;       // 4 Mi elements
    const size_t eW = (size_t)N_EMBD * N_EMBD;      // 1 Mi elements

    __bf16* xb  = (__bf16*)d_ws;
    __bf16* wqb = xb  + eX;
    __bf16* wkb = wqb + eW;
    __bf16* wvb = wkb + eW;
    __bf16* wpb = wvb + eW;
    __bf16* Qb  = wpb + eW;
    __bf16* Kb  = Qb  + eX;
    __bf16* Vtb = Kb  + eX;
    __bf16* Ob  = Vtb + eX;
    // total workspace: 24 Mi bf16 = 48 MB

    cvt_bf16<<<2048, 256, 0, stream>>>(x,  xb,  (int)eX);
    cvt_bf16<<<512,  256, 0, stream>>>(Wq, wqb, (int)eW);
    cvt_bf16<<<512,  256, 0, stream>>>(Wk, wkb, (int)eW);
    cvt_bf16<<<512,  256, 0, stream>>>(Wv, wvb, (int)eW);
    cvt_bf16<<<512,  256, 0, stream>>>(Wp, wpb, (int)eW);

    dim3 gg(MROWS / 256, N_EMBD / 64);               // 16 x 16
    gemm1024<<<gg, 256, 0, stream>>>(xb, wqb, Qb,  nullptr, 0);
    gemm1024<<<gg, 256, 0, stream>>>(xb, wkb, Kb,  nullptr, 0);
    gemm1024<<<gg, 256, 0, stream>>>(xb, wvb, Vtb, nullptr, 1);

    attn_flash<<<dim3(SEQ / 16 / 8, BATCH * N_HEAD), 256, 0, stream>>>(Qb, Kb, Vtb, Ob);

    gemm1024<<<gg, 256, 0, stream>>>(Ob, wpb, d_out, bp, 2);
}
